// AttnDecoderRNN_32959579030360
// MI455X (gfx1250) — matrix operand order repaired and checked by reading, not compiled, not run on hardware
//
#include <hip/hip_runtime.h>
#include <hip/hip_bf16.h>

#define Bdim 256
#define Ldim 64
#define Hdim 1024
#define INdim 512
#define Vdim 32000

typedef __attribute__((ext_vector_type(16))) __bf16 v16bf;
typedef __attribute__((ext_vector_type(8)))  __bf16 v8bf;
typedef __attribute__((ext_vector_type(8)))  float  v8f;
typedef __attribute__((ext_vector_type(4)))  unsigned int u32x4;
typedef __attribute__((ext_vector_type(8)))  int    i32x8;
typedef __attribute__((ext_vector_type(4)))  int    i32x4;

__device__ __forceinline__ __bf16 to_bf16(float f) { return (__bf16)f; }

__device__ __forceinline__ v16bf cvt_frag(float4 a, float4 b, float4 c, float4 d) {
  v16bf r;
  r[0]  = to_bf16(a.x); r[1]  = to_bf16(a.y); r[2]  = to_bf16(a.z); r[3]  = to_bf16(a.w);
  r[4]  = to_bf16(b.x); r[5]  = to_bf16(b.y); r[6]  = to_bf16(b.z); r[7]  = to_bf16(b.w);
  r[8]  = to_bf16(c.x); r[9]  = to_bf16(c.y); r[10] = to_bf16(c.z); r[11] = to_bf16(c.w);
  r[12] = to_bf16(d.x); r[13] = to_bf16(d.y); r[14] = to_bf16(d.z); r[15] = to_bf16(d.w);
  return r;
}

__device__ __forceinline__ v8bf cvt8(float4 a, float4 b) {
  v8bf o;
  o[0] = to_bf16(a.x); o[1] = to_bf16(a.y); o[2] = to_bf16(a.z); o[3] = to_bf16(a.w);
  o[4] = to_bf16(b.x); o[5] = to_bf16(b.y); o[6] = to_bf16(b.z); o[7] = to_bf16(b.w);
  return o;
}

// bf16 fragment gather from LDS. Per CDNA5 ISA 16-bit A layout: lane holds
// row (lane&15); low half-wave K = {0..7,16..23}, high half-wave K += 8.
// p = row base + koff; reads 8 halves at +0 and 8 at +16.
__device__ __forceinline__ v16bf lds_frag_bf16(const __bf16* p) {
  v8bf lo = *(const v8bf*)(p);
  v8bf hi = *(const v8bf*)(p + 16);
  return __builtin_shufflevector(lo, hi, 0, 1, 2, 3, 4, 5, 6, 7,
                                 8, 9, 10, 11, 12, 13, 14, 15);
}

// Issue a TDM DMA: 2-D fp32 tile [256 rows x 32 cols] starting at gsrc
// (row pitch = ldk floats) into LDS at byte offset lds_off, with TDM padding
// of 4 dwords after every 32 dwords -> 36-dword LDS row pitch.
// D# layout per CDNA5 ISA ch.8 (groups 0/1; trailing groups zero for 2-D).
__device__ __forceinline__ void tdm_load_panel(const float* gsrc, unsigned lds_off, int ldk) {
  unsigned long long ga = (unsigned long long)(size_t)gsrc;
  u32x4 g0;
  g0[0] = 1u;                                        // count=1, user mode
  g0[1] = lds_off;                                   // lds_addr (bytes)
  g0[2] = (unsigned)(ga & 0xffffffffull);            // global_addr[31:0]
  g0[3] = (unsigned)((ga >> 32) & 0x01ffffffull) | (2u << 30);  // addr[56:32] | type=2
  i32x8 g1;
  g1[0] = (int)((2u << 16) | (1u << 20) | (4u << 22) | (3u << 25));
          // data_size=4B | pad_enable | pad_interval=32dw | pad_amount=4dw
  g1[1] = (int)(((unsigned)ldk & 0xffffu) << 16);    // tensor_dim0 lo16
  g1[2] = (int)((((unsigned)ldk >> 16) & 0xffffu) | (256u << 16)); // dim0 hi | dim1 lo
  g1[3] = (int)(32u << 16);                          // dim1 hi=0 | tile_dim0=32
  g1[4] = 256;                                       // tile_dim1=256, tile_dim2=0
  g1[5] = ldk;                                       // tensor_dim0_stride lo32
  g1[6] = 0;
  g1[7] = 0;
  i32x4 z4 = {0, 0, 0, 0};
  i32x8 z8 = {0, 0, 0, 0, 0, 0, 0, 0};
  __builtin_amdgcn_tensor_load_to_lds(g0, g1, z4, z4, z8, 0);
}

// ---------------------------------------------------------------------------
// C[256, N] = A[256, K] @ W[N, K]^T + bias.  bf16 WMMA, f32 accumulate.
// Per K-step: TDM DMA stages the fp32 A panel (double-buffered, prefetched one
// step ahead); ONE shared conversion pass (1 thread = 1 row) produces a bf16
// panel (pitch 40 halves -> conflict-free fragment reads); the WMMA loop is
// then pure ds_load_b128 -> v_wmma with no VALU touching fragment registers
// (kills the WMMA->VALU WAR nops and 7/8 of the conversions).
// W streamed from global with one-iteration register prefetch; all 16 M-tiles
// live as accumulators so W is read from HBM exactly once.
// Dynamic LDS: 2*256*36 fp32 + 256*40 bf16 = 94208 bytes.
// ---------------------------------------------------------------------------
__global__ void __launch_bounds__(256) wmma_gemm_bias(
    const float* __restrict__ A,
    const float* __restrict__ W, long ldw,
    const float* __restrict__ bias,
    float* __restrict__ C, long ldc, int K) {
  extern __shared__ float smem[];
  float*  A0 = smem;                            // fp32 TDM panel, 36-dword pitch
  float*  A1 = smem + 256 * 36;
  __bf16* Ab = (__bf16*)(smem + 2 * 256 * 36);  // bf16 panel, 40-half pitch

  const int tid  = threadIdx.x;
  const int wave = tid >> 5;
  const int lane = tid & 31;
  const int n0   = blockIdx.x * 128 + wave * 16;
  const int col  = lane & 15;
  const int rb   = (lane >> 4) << 3;
  const int koff = (lane >> 4) << 3;

  v8f acc[16] = {};
  const float* Wp = W + (long)(n0 + col) * ldw + koff;

  // prefetch raw B fragment for k0 = 0
  float4 w0 = *(const float4*)(Wp);
  float4 w1 = *(const float4*)(Wp + 4);
  float4 w2 = *(const float4*)(Wp + 16);
  float4 w3 = *(const float4*)(Wp + 20);

  const unsigned lds0 = (unsigned)(size_t)A0;
  const unsigned lds1 = (unsigned)(size_t)A1;
  if (wave == 0) tdm_load_panel(A, lds0, K);

  const int nk = K >> 5;
  for (int i = 0; i < nk; ++i) {
    if (wave == 0) __builtin_amdgcn_s_wait_tensorcnt(0);
    __syncthreads();        // fp32 panel i ready; bf16 panel reads (i-1) done
    if (wave == 0 && i + 1 < nk)
      tdm_load_panel(A + (i + 1) * 32, (i & 1) ? lds0 : lds1, K);  // overlap DMA

    // Shared conversion pass: thread tid converts row tid (32 floats -> bf16).
    {
      const float* srow = ((i & 1) ? A1 : A0) + tid * 36;
      __bf16* drow = Ab + tid * 40;
#pragma unroll
      for (int c = 0; c < 32; c += 8) {
        float4 f0 = *(const float4*)(srow + c);
        float4 f1 = *(const float4*)(srow + c + 4);
        *(v8bf*)(drow + c) = cvt8(f0, f1);
      }
    }

    v16bf bf = cvt_frag(w0, w1, w2, w3);  // consume loads issued last iteration
    if (i + 1 < nk) {                     // prefetch next raw B fragment
      const float* Wn = Wp + (i + 1) * 32;
      w0 = *(const float4*)(Wn);
      w1 = *(const float4*)(Wn + 4);
      w2 = *(const float4*)(Wn + 16);
      w3 = *(const float4*)(Wn + 20);
    }

    __syncthreads();                      // bf16 panel ready for all waves
    const __bf16* Ap = Ab + col * 40 + koff;
#pragma unroll
    for (int t = 0; t < 16; ++t) {
      v16bf af = lds_frag_bf16(Ap + (t << 4) * 40);
      acc[t] = __builtin_amdgcn_wmma_f32_16x16x32_bf16(
          false, af, false, bf, (short)0, acc[t], false, false);
    }
  }

  const float bv = bias[n0 + col];
#pragma unroll
  for (int t = 0; t < 16; ++t) {
#pragma unroll
    for (int j = 0; j < 8; ++j) {
      C[(long)((t << 4) + rb + j) * ldc + n0 + col] = acc[t][j] + bv;
    }
  }
}

// ---------------------------------------------------------------------------
// Fused attention scores: scores[b,l] = sum_n v[n]*tanh(u[b,n] + enc.We^T).
// Grid = (L, B/16). The 16x2048 enc panel is converted to bf16 and staged in
// LDS once (pitch 2056 halves -> conflict-free fragment reads); W_e fragments
// stream from global (L2-resident) with one-ahead register prefetch.
// Dynamic LDS: 16 floats + 16*2056 bf16 = 65856 bytes.
// ---------------------------------------------------------------------------
__global__ void __launch_bounds__(256) attn_scores_kernel(
    const float* __restrict__ enc, const float* __restrict__ Wattn,
    const float* __restrict__ u,   const float* __restrict__ v_attn,
    float* __restrict__ scores) {
  extern __shared__ float smemf[];
  float* sred = smemf;
  __bf16* Ab  = (__bf16*)(smemf + 16);
  const int APITCH = 2056;

  const int l    = blockIdx.x;
  const int b0   = blockIdx.y << 4;
  const int tid  = threadIdx.x;
  const int wave = tid >> 5;
  const int lane = tid & 31;
  const int col  = lane & 15;
  const int koff = (lane >> 4) << 3;

  if (tid < 16) sred[tid] = 0.f;

  // Stage enc rows b0..b0+15 (2048 floats each) as bf16 into LDS.
  {
    const int row = tid >> 4;
    const int seg = tid & 15;
    const float* src = enc + ((long)l * Bdim + b0 + row) * (2 * Hdim) + seg * 128;
    __bf16* dst = Ab + row * APITCH + seg * 128;
#pragma unroll 4
    for (int c = 0; c < 128; c += 8) {
      float4 f0 = *(const float4*)(src + c);
      float4 f1 = *(const float4*)(src + c + 4);
      *(v8bf*)(dst + c) = cvt8(f0, f1);
    }
  }
  __syncthreads();

  const float* We = Wattn + Hdim;        // W_e = W_attn[:, H:3H]
  const __bf16* arow = Ab + col * APITCH + koff;

  v8f acc[8] = {};
  const int NKK = (2 * Hdim / 32) * 8;   // 512 flattened (k, nt) steps

  // prefetch (kk=0): nt=0, k0=0
  const float* bp = We + (long)(wave * 128 + col) * (3 * Hdim) + koff;
  float4 w0 = *(const float4*)(bp);
  float4 w1 = *(const float4*)(bp + 4);
  float4 w2 = *(const float4*)(bp + 16);
  float4 w3 = *(const float4*)(bp + 20);
  v16bf af = lds_frag_bf16(arow);

#pragma unroll 8
  for (int kk = 0; kk < NKK; ++kk) {
    const int nt = kk & 7;
    v16bf bf = cvt_frag(w0, w1, w2, w3);
    if (kk + 1 < NKK) {                  // prefetch next (k, nt) fragment
      const int kn  = kk + 1;
      const int k0n = (kn >> 3) << 5;
      const float* bn = We + (long)(wave * 128 + ((kn & 7) << 4) + col) * (3 * Hdim)
                        + k0n + koff;
      w0 = *(const float4*)(bn);
      w1 = *(const float4*)(bn + 4);
      w2 = *(const float4*)(bn + 16);
      w3 = *(const float4*)(bn + 20);
    }
    acc[nt] = __builtin_amdgcn_wmma_f32_16x16x32_bf16(
        false, af, false, bf, (short)0, acc[nt], false, false);
    if (nt == 7 && kk + 1 < NKK)
      af = lds_frag_bf16(arow + (((kk + 1) >> 3) << 5));
  }

  const int rbr = (lane >> 4) << 3;
  float part[8];
#pragma unroll
  for (int j = 0; j < 8; ++j) part[j] = 0.f;

#pragma unroll
  for (int nt = 0; nt < 8; ++nt) {
    const int n = wave * 128 + (nt << 4) + col;
    const float vn = v_attn[n];
#pragma unroll
    for (int j = 0; j < 8; ++j) {
      const float e = tanhf(acc[nt][j] + u[(long)(b0 + rbr + j) * Hdim + n]);
      part[j] += vn * e;
    }
  }

#pragma unroll
  for (int off = 8; off >= 1; off >>= 1) {
#pragma unroll
    for (int j = 0; j < 8; ++j) part[j] += __shfl_xor(part[j], off, 32);
  }

  __syncthreads();
  if ((lane & 15) == 0) {
#pragma unroll
    for (int j = 0; j < 8; ++j) atomicAdd(&sred[rbr + j], part[j]);
  }
  __syncthreads();
  if (tid < 16)
    scores[(long)(b0 + tid) * Ldim + l] = sred[tid];
}

// ---------------------------------------------------------------------------
// Mask + softmax over L, emit attn_weights, context = attn @ enc.
// ---------------------------------------------------------------------------
__global__ void __launch_bounds__(256) softmax_weighted_kernel(
    const float* __restrict__ scores, const unsigned char* __restrict__ mask,
    const float* __restrict__ enc, float* __restrict__ weighted,
    float* __restrict__ attn_out) {
  const int b = blockIdx.x;
  const int tid = threadIdx.x;
  __shared__ float sw[Ldim];

  if (tid < Ldim) {
    float v = scores[(long)b * Ldim + tid];
    if (mask[(long)b * Ldim + tid]) v = -1e10f;
    sw[tid] = v;
  }
  __syncthreads();
  if (tid == 0) {
    float mx = sw[0];
    for (int i = 1; i < Ldim; ++i) mx = fmaxf(mx, sw[i]);
    float sum = 0.f;
    for (int i = 0; i < Ldim; ++i) { float e = __expf(sw[i] - mx); sw[i] = e; sum += e; }
    const float inv = 1.f / sum;
    for (int i = 0; i < Ldim; ++i) sw[i] *= inv;
  }
  __syncthreads();
  if (tid < Ldim) attn_out[(long)b * Ldim + tid] = sw[tid];

  for (int d = tid; d < 2 * Hdim; d += blockDim.x) {
    float acc = 0.f;
    for (int l = 0; l < Ldim; ++l)
      acc += sw[l] * enc[((long)l * Bdim + b) * (2 * Hdim) + d];
    weighted[(long)b * (2 * Hdim) + d] = acc;
  }
}

// x = [input_emb | weighted]  (B x 2560)
__global__ void build_x_kernel(const float* __restrict__ emb,
                               const float* __restrict__ weighted,
                               float* __restrict__ x) {
  const int i = blockIdx.x * blockDim.x + threadIdx.x;
  if (i >= Bdim * (INdim + 2 * Hdim)) return;
  const int b = i / (INdim + 2 * Hdim);
  const int c = i % (INdim + 2 * Hdim);
  x[i] = (c < INdim) ? emb[(long)b * INdim + c]
                     : weighted[(long)b * 2 * Hdim + (c - INdim)];
}

// GRU cell elementwise; also writes h_new into out_in[:, :H]
__global__ void gru_kernel(const float* __restrict__ gi, const float* __restrict__ gh,
                           const float* __restrict__ h, float* __restrict__ hnew_out,
                           float* __restrict__ out_in) {
  const int i = blockIdx.x * blockDim.x + threadIdx.x;
  if (i >= Bdim * Hdim) return;
  const int b = i / Hdim, c = i % Hdim;
  const long base = (long)b * 3 * Hdim;
  const float r = 1.f / (1.f + __expf(-(gi[base + c] + gh[base + c])));
  const float z = 1.f / (1.f + __expf(-(gi[base + Hdim + c] + gh[base + Hdim + c])));
  const float n = tanhf(gi[base + 2 * Hdim + c] + r * gh[base + 2 * Hdim + c]);
  const float hn = (1.f - z) * n + z * h[i];
  hnew_out[i] = hn;
  out_in[(long)b * (3 * Hdim + INdim) + c] = hn;
}

// out_in[:, H:3H] = weighted ; out_in[:, 3H:3H+IN] = input_emb
__global__ void build_outin_kernel(const float* __restrict__ weighted,
                                   const float* __restrict__ emb,
                                   float* __restrict__ out_in) {
  const int i = blockIdx.x * blockDim.x + threadIdx.x;
  if (i >= Bdim * (2 * Hdim + INdim)) return;
  const int b = i / (2 * Hdim + INdim);
  const int c = i % (2 * Hdim + INdim);
  out_in[(long)b * (3 * Hdim + INdim) + Hdim + c] =
      (c < 2 * Hdim) ? weighted[(long)b * 2 * Hdim + c]
                     : emb[(long)b * INdim + (c - 2 * Hdim)];
}

extern "C" void kernel_launch(void* const* d_in, const int* in_sizes, int n_in,
                              void* d_out, int out_size, void* d_ws, size_t ws_size,
                              hipStream_t stream) {
  const float* input_emb = (const float*)d_in[0];
  const float* hidden    = (const float*)d_in[1];
  const float* enc       = (const float*)d_in[2];
  const unsigned char* mask = (const unsigned char*)d_in[3];
  const float* W_attn    = (const float*)d_in[4];
  const float* b_attn    = (const float*)d_in[5];
  const float* v_attn    = (const float*)d_in[6];
  const float* W_ih      = (const float*)d_in[7];
  const float* W_hh      = (const float*)d_in[8];
  const float* b_ih      = (const float*)d_in[9];
  const float* b_hh      = (const float*)d_in[10];
  const float* W_out     = (const float*)d_in[11];
  const float* b_out     = (const float*)d_in[12];

  float* out    = (float*)d_out;
  float* logits = out;                               // [B, V]
  float* hnew   = out + (long)Bdim * Vdim;           // [B, H]
  float* attn_w = hnew + (long)Bdim * Hdim;          // [B, L]

  float* ws       = (float*)d_ws;
  float* u        = ws;  ws += (long)Bdim * Hdim;
  float* scores   = ws;  ws += (long)Bdim * Ldim;
  float* weighted = ws;  ws += (long)Bdim * 2 * Hdim;
  float* x        = ws;  ws += (long)Bdim * (INdim + 2 * Hdim);
  float* gi       = ws;  ws += (long)Bdim * 3 * Hdim;
  float* gh       = ws;  ws += (long)Bdim * 3 * Hdim;
  float* out_in   = ws;  ws += (long)Bdim * (3 * Hdim + INdim);

  const size_t gemm_lds = 2 * 256 * 36 * sizeof(float) + 256 * 40 * 2;  // 94208 B
  const size_t attn_lds = 16 * sizeof(float) + 16 * 2056 * 2;           // 65856 B

  // 1) u = h @ W_h^T + b_attn           (W_h = W_attn[:, :H])
  wmma_gemm_bias<<<Hdim / 128, 256, gemm_lds, stream>>>(
      hidden, W_attn, 3 * Hdim, b_attn, u, Hdim, Hdim);

  // 2) scores[b,l] = v . tanh(u + enc @ W_e^T)
  attn_scores_kernel<<<dim3(Ldim, Bdim / 16), 256, attn_lds, stream>>>(
      enc, W_attn, u, v_attn, scores);

  // 3) softmax + mask -> attn_weights; context = attn @ enc
  softmax_weighted_kernel<<<Bdim, 256, 0, stream>>>(
      scores, mask, enc, weighted, attn_w);

  // 4) x = [emb | context]
  build_x_kernel<<<(Bdim * (INdim + 2 * Hdim) + 255) / 256, 256, 0, stream>>>(
      input_emb, weighted, x);

  // 5) gi = x @ W_ih^T + b_ih
  wmma_gemm_bias<<<(3 * Hdim) / 128, 256, gemm_lds, stream>>>(
      x, W_ih, INdim + 2 * Hdim, b_ih, gi, 3 * Hdim, INdim + 2 * Hdim);

  // 6) gh = h @ W_hh^T + b_hh
  wmma_gemm_bias<<<(3 * Hdim) / 128, 256, gemm_lds, stream>>>(
      hidden, W_hh, Hdim, b_hh, gh, 3 * Hdim, Hdim);

  // 7) GRU elementwise -> h_new (output + out_in[:, :H])
  gru_kernel<<<(Bdim * Hdim + 255) / 256, 256, 0, stream>>>(
      gi, gh, hidden, hnew, out_in);

  // 8) out_in[:, H:] = [context | emb]
  build_outin_kernel<<<(Bdim * (2 * Hdim + INdim) + 255) / 256, 256, 0, stream>>>(
      weighted, input_emb, out_in);

  // 9) logits = out_in @ W_out^T + b_out   (W_out streamed from HBM once)
  wmma_gemm_bias<<<Vdim / 128, 256, gemm_lds, stream>>>(
      out_in, W_out, 3 * Hdim + INdim, b_out, logits, Vdim, 3 * Hdim + INdim);
}
